// HyperStyleHWR_31061203485207
// MI455X (gfx1250) — matrix operand order repaired and checked by reading, not compiled, not run on hardware
//
#include <hip/hip_runtime.h>

// CDNA5 / gfx1250, wave32. WMMA f32 = f16 x f16 + f32 (16x16x32).
typedef __attribute__((ext_vector_type(16))) _Float16 v16h;
typedef __attribute__((ext_vector_type(8)))  float    v8f;

#define WAVE 32

// ---- fragment loaders -----------------------------------------------------
// A 16x32 f16 layout (per lane: row = lane&15): K = hi*16 + j
__device__ __forceinline__ v16h frag_a_fast(const float* __restrict__ aRow, int k0, int hi)
{
    v16h a;
    int ka0 = k0 + (hi << 4);
#pragma unroll
    for (int j = 0; j < 8; ++j) {
        a[j]     = (_Float16)aRow[ka0 + j];
        a[j + 8] = (_Float16)aRow[ka0 + 8 + j];
    }
    return a;
}
// B 32x16 f16 layout (per lane: col = lane&15): K = hi*16 + j
__device__ __forceinline__ v16h frag_b_fast(const float* __restrict__ bBase, int k0, int hi,
                                            int ldb, int transB)
{
    v16h b;
    int kb0 = k0 + (hi << 4);
    if (transB) {
#pragma unroll
        for (int j = 0; j < 16; ++j) b[j] = (_Float16)bBase[kb0 + j];
    } else {
#pragma unroll
        for (int j = 0; j < 16; ++j) b[j] = (_Float16)bBase[(kb0 + j) * ldb];
    }
    return b;
}

// ---------------------------------------------------------------------------
// Generic batched WMMA GEMM:  C[batch] = act( A[batch] * B[batch'] + bias )
//   A: f32 [M,K] row-major, batch stride aS (elements)
//   B: f32 [K,N] row-major (or [N,K] if transB), batch index = modB ? batch%modB : batch
//   bias: mode 0 none, 1 per-row(M), 2 per-col(N); batch index same rule as B
//   act: 0 none, 1 relu
// Each wave computes one 16x16 tile. Interior tiles run a x2-unrolled fast
// loop (unconditional clause-batched loads, two back-to-back WMMAs); edge
// tiles use clamped addresses + select (branchless, full EXEC for WMMA).
// ---------------------------------------------------------------------------
__global__ void k_wmma_gemm(
    const float* __restrict__ A, long long aS, int lda,
    const float* __restrict__ Bm, long long bS, int ldb, int transB, int modB,
    float* __restrict__ C, long long cS, int ldc,
    const float* __restrict__ bias, long long biasS, int biasMode,
    int M, int N, int K, int act,
    int tilesPerBatch, int tilesN, long long totalTiles)
{
    long long tile = (long long)blockIdx.x * (blockDim.x / WAVE) + (threadIdx.x / WAVE);
    if (tile >= totalTiles) return;               // uniform per wave
    int lane  = threadIdx.x & (WAVE - 1);
    int row16 = lane & 15;
    int hi    = lane >> 4;                        // 0: lanes 0-15, 1: lanes 16-31

    long long batch = tile / tilesPerBatch;
    int r  = (int)(tile % tilesPerBatch);
    int m0 = (r / tilesN) * 16;
    int n0 = (r % tilesN) * 16;
    long long bb = modB ? (batch % modB) : batch;

    const float* Ab = A  + batch * aS;
    const float* Bb = Bm + bb    * bS;
    float*       Cb = C  + batch * cS;
    const float* biasB = bias + bb * biasS;

    int am = m0 + row16;        // A: row fixed per lane
    int bn = n0 + row16;        // B: col fixed per lane
    bool fullMN = (m0 + 16 <= M) && (n0 + 16 <= N);   // wave-uniform

    // Per-lane base pointers (row/col clamped; clamp is a no-op on the fast path).
    const float* aRow  = Ab + (long long)(am < M ? am : (M - 1)) * lda;
    const float* bBase = transB ? (Bb + (long long)(bn < N ? bn : (N - 1)) * ldb)
                                : (Bb + (bn < N ? bn : (N - 1)));

    v8f acc = {};
    int kMain = fullMN ? (K & ~31) : 0;   // fast region (wave-uniform)
    int k0 = 0;

    // ---- fast main loop, unrolled x2: one load clause, two WMMAs ----
    for (; k0 + 64 <= kMain; k0 += 64) {
        __builtin_prefetch(aRow + k0 + 128, 0, 0);   // global_prefetch_b8
        v16h a0 = frag_a_fast(aRow, k0, hi);
        v16h b0 = frag_b_fast(bBase, k0, hi, ldb, transB);
        v16h a1 = frag_a_fast(aRow, k0 + 32, hi);
        v16h b1 = frag_b_fast(bBase, k0 + 32, hi, ldb, transB);
        acc = __builtin_amdgcn_wmma_f32_16x16x32_f16(false, a0, false, b0,
                                                     (short)0, acc, false, false);
        acc = __builtin_amdgcn_wmma_f32_16x16x32_f16(false, a1, false, b1,
                                                     (short)0, acc, false, false);
    }
    if (k0 < kMain) {
        v16h a0 = frag_a_fast(aRow, k0, hi);
        v16h b0 = frag_b_fast(bBase, k0, hi, ldb, transB);
        acc = __builtin_amdgcn_wmma_f32_16x16x32_f16(false, a0, false, b0,
                                                     (short)0, acc, false, false);
        k0 += 32;
    }
    // ---- guarded remainder / edge tiles: clamped addresses + select ----
    for (; k0 < K; k0 += 32) {
        v16h a, b;
#pragma unroll
        for (int j = 0; j < 16; ++j) {
            int ka  = k0 + (hi << 4) + j;
            float av = aRow[ka < K ? ka : 0];
            a[j] = (_Float16)((am < M && ka < K) ? av : 0.0f);
            int kb  = k0 + (hi << 4) + j;
            int kbc = kb < K ? kb : 0;
            float bv = transB ? bBase[kbc] : bBase[kbc * ldb];
            b[j] = (_Float16)((bn < N && kb < K) ? bv : 0.0f);
        }
        acc = __builtin_amdgcn_wmma_f32_16x16x32_f16(false, a, false, b,
                                                     (short)0, acc, false, false);
    }

    // C/D layout: VGPR v -> M = m0 + v + hi*8, N = n0 + (lane&15)
#pragma unroll
    for (int v = 0; v < 8; ++v) {
        int m = m0 + v + (hi << 3);
        int n = n0 + row16;
        if (m < M && n < N) {
            float val = acc[v];
            if (biasMode == 1)      val += biasB[m];
            else if (biasMode == 2) val += biasB[n];
            if (act) val = fmaxf(val, 0.0f);
            Cb[(long long)m * ldc + n] = val;
        }
    }
}

// ---------------------------------------------------------------------------
// conv0 (1->64, 3x3, pad1) + relu + maxpool 2x2/2: in[2,1,64,512] -> out[2,64,32,256]
// ---------------------------------------------------------------------------
__global__ void k_conv0_relu_pool(const float* __restrict__ in,
                                  const float* __restrict__ w,
                                  const float* __restrict__ bsv,
                                  float* __restrict__ out)
{
    int idx = blockIdx.x * blockDim.x + threadIdx.x;
    const int TOT = 2 * 64 * 32 * 256;
    if (idx >= TOT) return;
    int xo = idx % 256; int t = idx / 256;
    int yo = t % 32; t /= 32;
    int c  = t % 64; int b = t / 64;
    float m = -1e30f;
    for (int dy = 0; dy < 2; ++dy)
        for (int dx = 0; dx < 2; ++dx) {
            int y = yo * 2 + dy, x = xo * 2 + dx;
            float acc = bsv[c];
            for (int ky = 0; ky < 3; ++ky)
                for (int kx = 0; kx < 3; ++kx) {
                    int iy = y + ky - 1, ix = x + kx - 1;
                    if (iy >= 0 && iy < 64 && ix >= 0 && ix < 512)
                        acc += w[c * 9 + ky * 3 + kx] * in[(b * 64 + iy) * 512 + ix];
                }
            m = fmaxf(m, acc);
        }
    out[idx] = fmaxf(m, 0.0f);
}

// ---------------------------------------------------------------------------
// im2col (3x3) with channel grouping: cols[(b*G+g)][cl*9+kk][y*Wo+x]
// ---------------------------------------------------------------------------
__global__ void k_im2col(const float* __restrict__ x, float* __restrict__ cols,
                         int Bn, int C, int H, int W, int G, int pad, int Ho, int Wo)
{
    long long L = (long long)Ho * Wo;
    long long total = (long long)Bn * C * 9 * L;
    long long idx = (long long)blockIdx.x * blockDim.x + threadIdx.x;
    if (idx >= total) return;
    int Cg = C / G;
    int l  = (int)(idx % L); long long t = idx / L;
    int kk = (int)(t % 9);  t /= 9;
    int cl = (int)(t % Cg); t /= Cg;
    int g  = (int)(t % G);  int b = (int)(t / G);
    int y = l / Wo, xo = l % Wo;
    int iy = y + (kk / 3) - pad, ix = xo + (kk % 3) - pad;
    int ci = g * Cg + cl;
    float v = 0.0f;
    if (iy >= 0 && iy < H && ix >= 0 && ix < W)
        v = x[((long long)(b * C + ci) * H + iy) * W + ix];
    cols[idx] = v;
}

// ---------------------------------------------------------------------------
// maxpool 2x2, generic stride/pad (-inf padding == skip OOB taps)
// ---------------------------------------------------------------------------
__global__ void k_maxpool(const float* __restrict__ x, float* __restrict__ y,
                          int Bn, int C, int H, int W,
                          int sh, int sw, int ph, int pw, int Ho, int Wo)
{
    long long total = (long long)Bn * C * Ho * Wo;
    long long idx = (long long)blockIdx.x * blockDim.x + threadIdx.x;
    if (idx >= total) return;
    int xo = (int)(idx % Wo); long long t = idx / Wo;
    int yo = (int)(t % Ho);  t /= Ho;
    int c  = (int)(t % C);   int b = (int)(t / C);
    int iy0 = yo * sh - ph, ix0 = xo * sw - pw;
    float m = -1e30f;
    for (int dy = 0; dy < 2; ++dy)
        for (int dx = 0; dx < 2; ++dx) {
            int iy = iy0 + dy, ix = ix0 + dx;
            if (iy >= 0 && iy < H && ix >= 0 && ix < W)
                m = fmaxf(m, x[((long long)(b * C + c) * H + iy) * W + ix]);
        }
    y[idx] = m;
}

// ---------------------------------------------------------------------------
// GroupNorm (biased var) + affine + relu, in place. One block per (b, group).
// ---------------------------------------------------------------------------
__global__ void k_gn_relu(float* __restrict__ x, const float* __restrict__ sc,
                          const float* __restrict__ bi, int C, int HW, int groups)
{
    int g = blockIdx.x % groups;
    int b = blockIdx.x / groups;
    int Cg = C / groups;
    long long n = (long long)Cg * HW;
    float* base = x + ((long long)b * C + (long long)g * Cg) * HW;
    __shared__ float ssum[256], ssq[256];
    int tid = threadIdx.x;
    float s1 = 0.0f, s2 = 0.0f;
    for (long long i = tid; i < n; i += 256) { float v = base[i]; s1 += v; s2 += v * v; }
    ssum[tid] = s1; ssq[tid] = s2;
    __syncthreads();
    for (int off = 128; off; off >>= 1) {
        if (tid < off) { ssum[tid] += ssum[tid + off]; ssq[tid] += ssq[tid + off]; }
        __syncthreads();
    }
    float inv_n = 1.0f / (float)n;
    float mean = ssum[0] * inv_n;
    float var  = ssq[0] * inv_n - mean * mean;
    float rstd = rsqrtf(var + 1e-5f);
    for (long long i = tid; i < n; i += 256) {
        int c = g * Cg + (int)(i / HW);
        float v = (base[i] - mean) * rstd * sc[c] + bi[c];
        base[i] = fmaxf(v, 0.0f);
    }
}

// ---------------------------------------------------------------------------
// hypernet layer-1: h1[b,g,h] = relu(style[b,:256] . W1[g,h,:256] + b1[g,h])
// ---------------------------------------------------------------------------
__global__ void k_hyper_h1(const float* __restrict__ style, const float* __restrict__ W1,
                           const float* __restrict__ b1, float* __restrict__ h1,
                           int G, int HD)
{
    int idx = blockIdx.x * blockDim.x + threadIdx.x;
    int total = 2 * G * HD;
    if (idx >= total) return;
    int h = idx % HD; int t = idx / HD;
    int g = t % G;    int b = t / G;
    const float* wrow = W1 + ((long long)g * HD + h) * 256;
    const float* st   = style + b * 256;
    float acc = b1[g * HD + h];
    for (int s = 0; s < 256; ++s) acc += st[s] * wrow[s];
    h1[idx] = fmaxf(acc, 0.0f);
}

// conv6 output [2,512,1,126] -> seq [126,2,512]
__global__ void k_seq_transpose(const float* __restrict__ x, float* __restrict__ seq)
{
    int idx = blockIdx.x * blockDim.x + threadIdx.x;
    const int TOT = 2 * 512 * 126;
    if (idx >= TOT) return;
    int t = idx % 126; int r = idx / 126;
    int c = r % 512;   int b = r / 512;
    seq[((long long)t * 2 + b) * 512 + c] = x[idx];
}

__global__ void k_add_vec(const float* __restrict__ a, const float* __restrict__ b,
                          float* __restrict__ o, int n)
{
    int i = blockIdx.x * blockDim.x + threadIdx.x;
    if (i < n) o[i] = a[i] + b[i];
}

// ---------------------------------------------------------------------------
// Persistent LSTM scan. One block per direction (block 0 fwd, block 1 rev).
// pre already contains x.Wih^T + bih + bhh. h,c,gates live in LDS.
// Gate order (i,f,g,o), torch convention. Writes into concat buffer halves.
// ---------------------------------------------------------------------------
__global__ void k_lstm_scan(const float* __restrict__ pre_f, const float* __restrict__ pre_r,
                            const float* __restrict__ whh_f, const float* __restrict__ whh_r,
                            float* __restrict__ out_f, float* __restrict__ out_r,
                            int T, int outStride)
{
    const bool rev = (blockIdx.x == 1);
    const float* pre = rev ? pre_r : pre_f;
    const float* Whh = rev ? whh_r : whh_f;
    float*       out = rev ? out_r : out_f;
    __shared__ float h[1024], c[1024], gates[4096]; // [2][512], [2][512], [2][2048]
    int tid = threadIdx.x; // 1024 threads
    h[tid] = 0.0f; c[tid] = 0.0f;
    __syncthreads();
    for (int step = 0; step < T; ++step) {
        int t = rev ? (T - 1 - step) : step;
        const float* p = pre + (long long)t * 4096;
        for (int e = tid; e < 4096; e += 1024) {
            int b = e >> 11, j = e & 2047;
            float acc = p[e];
            const float* wr = Whh + (long long)j * 512;
            const float* hb = h + b * 512;
            for (int k = 0; k < 512; ++k) acc += hb[k] * wr[k];
            gates[e] = acc;
        }
        __syncthreads();
        int b = tid >> 9, hh = tid & 511;
        float gi = gates[b * 2048 + hh];
        float gf = gates[b * 2048 + 512 + hh];
        float gg = gates[b * 2048 + 1024 + hh];
        float go = gates[b * 2048 + 1536 + hh];
        float si = 1.0f / (1.0f + expf(-gi));
        float sf = 1.0f / (1.0f + expf(-gf));
        float so = 1.0f / (1.0f + expf(-go));
        float cc = sf * c[tid] + si * tanhf(gg);
        float hn = so * tanhf(cc);
        c[tid] = cc; h[tid] = hn;
        out[((long long)t * 2 + b) * outStride + hh] = hn;
        __syncthreads();
    }
}

// ---------------------------------------------------------------------------
// Host-side orchestration
// ---------------------------------------------------------------------------
static void gemm(hipStream_t s,
                 const float* A, long long aS, int lda,
                 const float* B, long long bS, int ldb, int transB, int modB,
                 float* C, long long cS, int ldc,
                 const float* bias, long long biasS, int biasMode,
                 int M, int N, int K, int act, int batchCount)
{
    int tM = (M + 15) / 16, tN = (N + 15) / 16;
    long long total = (long long)batchCount * tM * tN;
    int blocks = (int)((total + 7) / 8);
    k_wmma_gemm<<<blocks, 256, 0, s>>>(A, aS, lda, B, bS, ldb, transB, modB,
                                       C, cS, ldc, bias, biasS, biasMode,
                                       M, N, K, act, tM * tN, tN, total);
}

static inline int blk(long long n) { return (int)((n + 255) / 256); }

extern "C" void kernel_launch(void* const* d_in, const int* in_sizes, int n_in,
                              void* d_out, int out_size, void* d_ws, size_t ws_size,
                              hipStream_t stream)
{
    (void)in_sizes; (void)n_in; (void)out_size; (void)ws_size;
    // ---- inputs (setup_inputs dict order, recursively) ----
    const float* input = (const float*)d_in[0];   // [2,1,64,512]
    const float* style = (const float*)d_in[1];   // [2,256]
    const float* w0 = (const float*)d_in[2];  const float* b0 = (const float*)d_in[3];
    const float* w2 = (const float*)d_in[4];  const float* b2 = (const float*)d_in[5];
    const float* gn2s = (const float*)d_in[6]; const float* gn2b = (const float*)d_in[7];
    const float* w4 = (const float*)d_in[8];  const float* b4 = (const float*)d_in[9];
    const float* gn4s = (const float*)d_in[10]; const float* gn4b = (const float*)d_in[11];
    const float* w6 = (const float*)d_in[12]; const float* b6 = (const float*)d_in[13];
    const float* gn6s = (const float*)d_in[14]; const float* gn6b = (const float*)d_in[15];
    const float* h1W1 = (const float*)d_in[16]; const float* h1b1 = (const float*)d_in[17];
    const float* h1W2 = (const float*)d_in[18]; const float* h1b2 = (const float*)d_in[19];
    const float* h2W1 = (const float*)d_in[20]; const float* h2b1 = (const float*)d_in[21];
    const float* h2W2 = (const float*)d_in[22]; const float* h2b2 = (const float*)d_in[23];
    const float* h3W1 = (const float*)d_in[24]; const float* h3b1 = (const float*)d_in[25];
    const float* h3W2 = (const float*)d_in[26]; const float* h3b2 = (const float*)d_in[27];
    const float* l0f_Wih = (const float*)d_in[28]; const float* l0f_Whh = (const float*)d_in[29];
    const float* l0f_bih = (const float*)d_in[30]; const float* l0f_bhh = (const float*)d_in[31];
    const float* l0r_Wih = (const float*)d_in[32]; const float* l0r_Whh = (const float*)d_in[33];
    const float* l0r_bih = (const float*)d_in[34]; const float* l0r_bhh = (const float*)d_in[35];
    const float* l1f_Wih = (const float*)d_in[36]; const float* l1f_Whh = (const float*)d_in[37];
    const float* l1f_bih = (const float*)d_in[38]; const float* l1f_bhh = (const float*)d_in[39];
    const float* l1r_Wih = (const float*)d_in[40]; const float* l1r_Whh = (const float*)d_in[41];
    const float* l1r_bih = (const float*)d_in[42]; const float* l1r_bhh = (const float*)d_in[43];
    const float* Wemb = (const float*)d_in[44]; const float* bemb = (const float*)d_in[45];
    float* outp = (float*)d_out;                  // [126,2,80] f32

    // ---- workspace arena (floats) ----
    float* w = (float*)d_ws; size_t off = 0;
    auto alloc = [&](size_t n) { float* p = w + off; off += n; return p; };
    float* x0    = alloc(2 * 64 * 32 * 256);      // conv0+pool out
    float* cols  = alloc(9437184);                // shared im2col buffer (max extent)
    float* h1buf = alloc(2 * 8 * 2376);           // hypernet hidden (max)
    float* gened = alloc(2 * 8 * 36928);          // hypernet generated w+b (max)
    float* a1    = alloc(2 * 128 * 32 * 256);     // hconv1 out
    float* x1    = alloc(2 * 128 * 16 * 128);     // pooled
    float* a2    = alloc(2 * 256 * 16 * 128);     // conv2 out (GN in place)
    float* a2b   = alloc(2 * 256 * 16 * 128);     // hconv2 out
    float* x2    = alloc(2 * 256 * 8 * 129);      // pooled
    float* a4    = alloc(2 * 512 * 8 * 129);      // conv4 out (GN in place)
    float* a5    = alloc(2 * 512 * 6 * 127);      // hconv3 out
    float* x3    = alloc(2 * 512 * 3 * 128);      // pooled
    float* a6    = alloc(2 * 512 * 1 * 126);      // conv6 out (GN in place)
    float* seq   = alloc(126 * 2 * 512);
    float* bsf   = alloc(2048);
    float* bsr   = alloc(2048);
    float* pref  = alloc(126 * 2 * 2048);
    float* prer  = alloc(126 * 2 * 2048);
    float* cat1  = alloc(126 * 2 * 1024);
    float* cat2  = alloc(126 * 2 * 1024);

    // ---- stage 0: conv0 + relu + pool ----
    k_conv0_relu_pool<<<blk(2LL * 64 * 32 * 256), 256, 0, stream>>>(input, w0, b0, x0);

    // ---- hyper1: generate weights, apply block-diag conv (G=2, g_in=32) ----
    k_hyper_h1<<<blk(2LL * 2 * 2376), 256, 0, stream>>>(style, h1W1, h1b1, h1buf, 2, 2376);
    gemm(stream, h1buf, 2376, 2376, h1W2, 18496LL * 2376, 2376, 1, 2,
         gened, 18496, 18496, h1b2, 18496, 2, 1, 18496, 2376, 0, 4);
    k_im2col<<<blk(2LL * 64 * 9 * 32 * 256), 256, 0, stream>>>(x0, cols, 2, 64, 32, 256, 2, 1, 32, 256);
    gemm(stream, gened, 18496, 288, cols, 288LL * 8192, 8192, 0, 0,
         a1, 64LL * 8192, 8192, gened + 18432, 18496, 1, 64, 8192, 288, 1, 4);
    k_maxpool<<<blk(2LL * 128 * 16 * 128), 256, 0, stream>>>(a1, x1, 2, 128, 32, 256, 2, 2, 0, 0, 16, 128);

    // ---- conv2 (128->256) + GN(32) + relu ----
    k_im2col<<<blk(2LL * 128 * 9 * 16 * 128), 256, 0, stream>>>(x1, cols, 2, 128, 16, 128, 1, 1, 16, 128);
    gemm(stream, w2, 0, 1152, cols, 1152LL * 2048, 2048, 0, 0,
         a2, 256LL * 2048, 2048, b2, 0, 1, 256, 2048, 1152, 0, 2);
    k_gn_relu<<<2 * 32, 256, 0, stream>>>(a2, gn2s, gn2b, 256, 2048, 32);

    // ---- hyper2 (G=4, g_in=64) ----
    k_hyper_h1<<<blk(2LL * 4 * 2340), 256, 0, stream>>>(style, h2W1, h2b1, h1buf, 4, 2340);
    gemm(stream, h1buf, 2340, 2340, h2W2, 36928LL * 2340, 2340, 1, 4,
         gened, 36928, 36928, h2b2, 36928, 2, 1, 36928, 2340, 0, 8);
    k_im2col<<<blk(2LL * 256 * 9 * 16 * 128), 256, 0, stream>>>(a2, cols, 2, 256, 16, 128, 4, 1, 16, 128);
    gemm(stream, gened, 36928, 576, cols, 576LL * 2048, 2048, 0, 0,
         a2b, 64LL * 2048, 2048, gened + 36864, 36928, 1, 64, 2048, 576, 1, 8);
    k_maxpool<<<blk(2LL * 256 * 8 * 129), 256, 0, stream>>>(a2b, x2, 2, 256, 16, 128, 2, 1, 0, 1, 8, 129);

    // ---- conv4 (256->512) + GN(32) + relu ----
    k_im2col<<<blk(2LL * 256 * 9 * 8 * 129), 256, 0, stream>>>(x2, cols, 2, 256, 8, 129, 1, 1, 8, 129);
    gemm(stream, w4, 0, 2304, cols, 2304LL * 1032, 1032, 0, 0,
         a4, 512LL * 1032, 1032, b4, 0, 1, 512, 1032, 2304, 0, 2);
    k_gn_relu<<<2 * 32, 256, 0, stream>>>(a4, gn4s, gn4b, 512, 1032, 32);

    // ---- hyper3 (G=8, g_in=64, pad 0) ----
    k_hyper_h1<<<blk(2LL * 8 * 1170), 256, 0, stream>>>(style, h3W1, h3b1, h1buf, 8, 1170);
    gemm(stream, h1buf, 1170, 1170, h3W2, 36928LL * 1170, 1170, 1, 8,
         gened, 36928, 36928, h3b2, 36928, 2, 1, 36928, 1170, 0, 16);
    k_im2col<<<blk(2LL * 512 * 9 * 6 * 127), 256, 0, stream>>>(a4, cols, 2, 512, 8, 129, 8, 0, 6, 127);
    gemm(stream, gened, 36928, 576, cols, 576LL * 762, 762, 0, 0,
         a5, 64LL * 762, 762, gened + 36864, 36928, 1, 64, 762, 576, 1, 16);
    k_maxpool<<<blk(2LL * 512 * 3 * 128), 256, 0, stream>>>(a5, x3, 2, 512, 6, 127, 2, 1, 0, 1, 3, 128);

    // ---- conv6 (512->512, pad 0) + GN(32) + relu ----
    k_im2col<<<blk(2LL * 512 * 9 * 1 * 126), 256, 0, stream>>>(x3, cols, 2, 512, 3, 128, 1, 0, 1, 126);
    gemm(stream, w6, 0, 4608, cols, 4608LL * 126, 126, 0, 0,
         a6, 512LL * 126, 126, b6, 0, 1, 512, 126, 4608, 0, 2);
    k_gn_relu<<<2 * 32, 256, 0, stream>>>(a6, gn6s, gn6b, 512, 126, 32);
    k_seq_transpose<<<blk(2LL * 512 * 126), 256, 0, stream>>>(a6, seq);

    // ---- BiLSTM layer 0 (I=512) ----
    k_add_vec<<<blk(2048), 256, 0, stream>>>(l0f_bih, l0f_bhh, bsf, 2048);
    k_add_vec<<<blk(2048), 256, 0, stream>>>(l0r_bih, l0r_bhh, bsr, 2048);
    gemm(stream, seq, 0, 512, l0f_Wih, 0, 512, 1, 0, pref, 0, 2048, bsf, 0, 2, 252, 2048, 512, 0, 1);
    gemm(stream, seq, 0, 512, l0r_Wih, 0, 512, 1, 0, prer, 0, 2048, bsr, 0, 2, 252, 2048, 512, 0, 1);
    k_lstm_scan<<<2, 1024, 0, stream>>>(pref, prer, l0f_Whh, l0r_Whh, cat1, cat1 + 512, 126, 1024);

    // ---- BiLSTM layer 1 (I=1024) ----
    k_add_vec<<<blk(2048), 256, 0, stream>>>(l1f_bih, l1f_bhh, bsf, 2048);
    k_add_vec<<<blk(2048), 256, 0, stream>>>(l1r_bih, l1r_bhh, bsr, 2048);
    gemm(stream, cat1, 0, 1024, l1f_Wih, 0, 1024, 1, 0, pref, 0, 2048, bsf, 0, 2, 252, 2048, 1024, 0, 1);
    gemm(stream, cat1, 0, 1024, l1r_Wih, 0, 1024, 1, 0, prer, 0, 2048, bsr, 0, 2, 252, 2048, 1024, 0, 1);
    k_lstm_scan<<<2, 1024, 0, stream>>>(pref, prer, l1f_Whh, l1r_Whh, cat2, cat2 + 512, 126, 1024);

    // ---- final projection: [252,1024] x Wemb^T + bemb -> [126,2,80] ----
    gemm(stream, cat2, 0, 1024, Wemb, 0, 1024, 1, 0, outp, 0, 80, bemb, 0, 2, 252, 80, 1024, 0, 1);
}
